// SSDLayer_44959717654935
// MI455X (gfx1250) — matrix operand order repaired and checked by reading, not compiled, not run on hardware
//
#include <hip/hip_runtime.h>
#include <hip/hip_bf16.h>

// Problem constants (from reference)
#define BSZ 2
#define TT  2048
#define HH  16
#define PP  64
#define NN  64
#define LCH 64            // chunk length
#define NCH (TT / LCH)    // 32 chunks
#define PCH 80            // bf16 LDS row pitch (160B = 5*32B: aligned + bank-spread)

typedef __attribute__((ext_vector_type(16))) __bf16 v16bf;
typedef __attribute__((ext_vector_type(8)))  __bf16 v8bf;
typedef __attribute__((ext_vector_type(8)))  float  v8f;

// Build A-operand fragment for v_wmma_f32_16x16x32_bf16 from a bf16 row in LDS.
// Lane layout (ISA 16-bit A 16x32): element a[2v+s] = row[KB + 16*(v>>2) + 8*half + 2*(v&3) + s]
//  -> two contiguous 8-element (16B) runs at row[KB + 8*half] and row[KB + 16 + 8*half].
__device__ __forceinline__ v16bf load_a_frag(const __bf16* row, int KB, int half) {
    const v8bf lo = *(const v8bf*)(row + KB + 8 * half);
    const v8bf hi = *(const v8bf*)(row + KB + 16 + 8 * half);
    return __builtin_shufflevector(lo, hi, 0, 1, 2, 3, 4, 5, 6, 7,
                                           8, 9, 10, 11, 12, 13, 14, 15);
}
// B-operand fragment: same k layout as A: element b[2v+s] = col_row[KB + 16*(v>>2) + 8*half + 2*(v&3) + s]
//  -> two contiguous 8-element (16B) runs at col_row[KB + 8*half] and col_row[KB + 16 + 8*half].
__device__ __forceinline__ v16bf load_b_frag(const __bf16* row, int KB, int half) {
    const v8bf lo = *(const v8bf*)(row + KB + 8 * half);
    const v8bf hi = *(const v8bf*)(row + KB + 16 + 8 * half);
    return __builtin_shufflevector(lo, hi, 0, 1, 2, 3, 4, 5, 6, 7,
                                           8, 9, 10, 11, 12, 13, 14, 15);
}

// ---------------------------------------------------------------------------
// Kernel 1: per (b,h,chunk)  [1024 blocks x 128 threads (4 waves)]
//   G = C@B^T (wmma) -> Gm = mask*decay*dt -> Y_intra = Gm@X (wmma)
//   S = X^T @ (csuf*B) (wmma) -> workspace; chunk decay -> workspace
// ---------------------------------------------------------------------------
__global__ __launch_bounds__(128) void ssd_chunk_kernel(
    const float* __restrict__ Xg, const float* __restrict__ delta,
    const float* __restrict__ Bg, const float* __restrict__ Cg,
    const float* __restrict__ A_log,
    float* __restrict__ Yout, float* __restrict__ Sws, float* __restrict__ Dws)
{
    __shared__ __attribute__((aligned(64))) __bf16 Cs16[LCH][PCH];   // [t][n]
    __shared__ __attribute__((aligned(64))) __bf16 Bs16[LCH][PCH];   // [t][n]
    __shared__ __attribute__((aligned(64))) __bf16 Xt16[PP][PCH];    // [p][t]   (X transposed)
    __shared__ __attribute__((aligned(64))) __bf16 Bwt16[NN][PCH];   // [n][j] = csuf[j]*B[j][n]
    __shared__ __attribute__((aligned(64))) __bf16 Gm16[LCH][PCH];   // [i][j]
    __shared__ float cum[LCH];
    __shared__ float dts[LCH];
    __shared__ float csuf[LCH];

    const int blk  = blockIdx.x;
    const int c    = blk % NCH;
    const int h    = (blk / NCH) % HH;
    const int b    = blk / (NCH * HH);
    const int t0   = c * LCH;
    const int tid  = threadIdx.x;
    const int lane = tid & 31;
    const int wv   = tid >> 5;      // wave 0..3
    const int half = lane >> 4;     // 0 or 1
    const int lm   = lane & 15;

    // ---- stage chunk operands into LDS as bf16 (coalesced global reads) ----
    for (int idx = tid; idx < LCH * PP; idx += 128) {
        int r = idx >> 6, col = idx & 63;
        float x = Xg[(((size_t)b * TT + t0 + r) * HH + h) * PP + col];
        Xt16[col][r] = (__bf16)x;                      // transposed
        Bs16[r][col] = (__bf16)Bg[((size_t)b * TT + t0 + r) * NN + col];
        Cs16[r][col] = (__bf16)Cg[((size_t)b * TT + t0 + r) * NN + col];
    }
    if (tid < LCH) dts[tid] = delta[((size_t)b * TT + t0 + tid) * HH + h];
    __syncthreads();

    const float alog = __expf(A_log[h]);
    if (tid < LCH) {
        float s = 0.f;
        for (int j = 0; j <= tid; ++j) s += dts[j];
        cum[tid] = -alog * s;                          // inclusive log-decay cumsum
    }
    __syncthreads();
    const float cum_last = cum[LCH - 1];
    if (tid < LCH) csuf[tid] = __expf(cum_last - cum[tid]) * dts[tid];
    __syncthreads();

    // Bwt[n][j] = csuf[j] * B[j][n]  (decay weighting moved off the WMMA path)
    for (int idx = tid; idx < LCH * NN; idx += 128) {
        int j = idx >> 6, n = idx & 63;
        Bwt16[n][j] = (__bf16)(csuf[j] * (float)Bs16[j][n]);
    }
    __syncthreads();

    const int tm   = wv;                 // each wave owns one 16-row block
    const int rowA = tm * 16 + lm;

    // per-lane invariant decay terms for the mask
    float cum_i[8];
#pragma unroll
    for (int r = 0; r < 8; ++r) cum_i[r] = cum[tm * 16 + r + 8 * half];

    // ---- GEMM 1: G = C @ B^T, branchless mask -> Gm16 ----
    for (int tn = 0; tn < 4; ++tn) {
        const int j  = tn * 16 + lm;
        const float cj = cum[j];
        const float dj = dts[j];
        v8f acc = {0.f, 0.f, 0.f, 0.f, 0.f, 0.f, 0.f, 0.f};
#pragma unroll
        for (int kk = 0; kk < 2; ++kk) {
            v16bf a  = load_a_frag(&Cs16[rowA][0], kk * 32, half);
            v16bf bb = load_b_frag(&Bs16[j][0],    kk * 32, half);   // Bop[k][j]=B[j][k]
            acc = __builtin_amdgcn_wmma_f32_16x16x32_bf16(
                false, a, false, bb, (short)0, acc, false, false);
        }
#pragma unroll
        for (int r = 0; r < 8; ++r) {
            int i = tm * 16 + r + 8 * half;
            float e = (j <= i) ? (cum_i[r] - cj) : -1e30f;   // exp -> 0 when masked
            Gm16[i][j] = (__bf16)(acc[r] * __expf(e) * dj);
        }
    }
    __syncthreads();

    // ---- GEMM 2: Y_intra = Gm @ X -> d_out ----
    for (int tn = 0; tn < 4; ++tn) {
        const int p = tn * 16 + lm;
        v8f acc = {0.f, 0.f, 0.f, 0.f, 0.f, 0.f, 0.f, 0.f};
#pragma unroll
        for (int kk = 0; kk < 2; ++kk) {
            v16bf a  = load_a_frag(&Gm16[rowA][0], kk * 32, half);
            v16bf bb = load_b_frag(&Xt16[p][0],    kk * 32, half);   // Bop[j][p]=Xt[p][j]
            acc = __builtin_amdgcn_wmma_f32_16x16x32_bf16(
                false, a, false, bb, (short)0, acc, false, false);
        }
#pragma unroll
        for (int r = 0; r < 8; ++r) {
            int i = tm * 16 + r + 8 * half;
            Yout[(((size_t)b * TT + t0 + i) * HH + h) * PP + p] = acc[r];
        }
    }

    // ---- GEMM 3: S[p][n] = sum_j X[j][p] * (csuf[j]*B[j][n]) -> workspace ----
    for (int tn = 0; tn < 4; ++tn) {
        const int n = tn * 16 + lm;
        v8f acc = {0.f, 0.f, 0.f, 0.f, 0.f, 0.f, 0.f, 0.f};
#pragma unroll
        for (int kk = 0; kk < 2; ++kk) {
            v16bf a  = load_a_frag(&Xt16[rowA][0], kk * 32, half);   // A[p][j]=Xt[p][j]
            v16bf bb = load_b_frag(&Bwt16[n][0],   kk * 32, half);   // Bop[j][n]=Bwt[n][j]
            acc = __builtin_amdgcn_wmma_f32_16x16x32_bf16(
                false, a, false, bb, (short)0, acc, false, false);
        }
#pragma unroll
        for (int r = 0; r < 8; ++r) {
            int p = tm * 16 + r + 8 * half;
            Sws[(size_t)blk * (PP * NN) + p * NN + n] = acc[r];
        }
    }
    if (tid == 0) Dws[blk] = __expf(cum_last);
}

// ---------------------------------------------------------------------------
// Kernel 2: chunk-level scan, per (b,h)  [32 blocks x 256 threads]
//   In-place: slot c of Sws becomes h_start of chunk c. Emits h_last.
// ---------------------------------------------------------------------------
__global__ __launch_bounds__(256) void ssd_scan_kernel(
    float* __restrict__ Sws, const float* __restrict__ Dws,
    float* __restrict__ Hlast)
{
    const int bh  = blockIdx.x;
    const int tid = threadIdx.x;
    float hreg[16];
#pragma unroll
    for (int k = 0; k < 16; ++k) hreg[k] = 0.f;

    for (int c = 0; c < NCH; ++c) {
        const float d = Dws[bh * NCH + c];
        const size_t base = ((size_t)bh * NCH + c) * (PP * NN);
#pragma unroll
        for (int k = 0; k < 16; ++k) {
            size_t idx = base + tid + k * 256;
            float s = Sws[idx];
            Sws[idx] = hreg[k];              // h_start for chunk c
            hreg[k] = d * hreg[k] + s;       // carry to next chunk
        }
    }
#pragma unroll
    for (int k = 0; k < 16; ++k)
        Hlast[(size_t)bh * (PP * NN) + tid + k * 256] = hreg[k];
}

// ---------------------------------------------------------------------------
// Kernel 3: inter-chunk output, per (b,h,chunk)  [1024 blocks x 128 threads]
//   Y[i,p] += exp(cum[i]) * (C @ h_start^T)[i,p]      (wmma)
// ---------------------------------------------------------------------------
__global__ __launch_bounds__(128) void ssd_inter_kernel(
    const float* __restrict__ delta, const float* __restrict__ Cg,
    const float* __restrict__ A_log, const float* __restrict__ Hst,
    float* __restrict__ Yout)
{
    __shared__ __attribute__((aligned(64))) __bf16 Cs16[LCH][PCH];   // [t][n]
    __shared__ __attribute__((aligned(64))) __bf16 Hs16[PP][PCH];    // [p][n]
    __shared__ float cum[LCH];
    __shared__ float dts[LCH];

    const int blk  = blockIdx.x;
    const int c    = blk % NCH;
    const int h    = (blk / NCH) % HH;
    const int b    = blk / (NCH * HH);
    const int t0   = c * LCH;
    const int tid  = threadIdx.x;
    const int lane = tid & 31;
    const int wv   = tid >> 5;
    const int half = lane >> 4;
    const int lm   = lane & 15;

    for (int idx = tid; idx < LCH * NN; idx += 128) {
        int r = idx >> 6, col = idx & 63;
        Cs16[r][col] = (__bf16)Cg[((size_t)b * TT + t0 + r) * NN + col];
        Hs16[r][col] = (__bf16)Hst[(size_t)blk * (PP * NN) + idx];
    }
    if (tid < LCH) dts[tid] = delta[((size_t)b * TT + t0 + tid) * HH + h];
    __syncthreads();
    const float alog = __expf(A_log[h]);
    if (tid < LCH) {
        float s = 0.f;
        for (int j = 0; j <= tid; ++j) s += dts[j];
        cum[tid] = -alog * s;
    }
    __syncthreads();

    const int tm   = wv;
    const int rowA = tm * 16 + lm;

    float esc[8];
#pragma unroll
    for (int r = 0; r < 8; ++r) esc[r] = __expf(cum[tm * 16 + r + 8 * half]);

    for (int tn = 0; tn < 4; ++tn) {
        const int p = tn * 16 + lm;
        v8f acc = {0.f, 0.f, 0.f, 0.f, 0.f, 0.f, 0.f, 0.f};
#pragma unroll
        for (int kk = 0; kk < 2; ++kk) {
            v16bf a  = load_a_frag(&Cs16[rowA][0], kk * 32, half);
            v16bf bb = load_b_frag(&Hs16[p][0],    kk * 32, half);   // Bop[n][p]=Hs[p][n]
            acc = __builtin_amdgcn_wmma_f32_16x16x32_bf16(
                false, a, false, bb, (short)0, acc, false, false);
        }
#pragma unroll
        for (int r = 0; r < 8; ++r) {
            int i = tm * 16 + r + 8 * half;
            size_t yi = (((size_t)b * TT + t0 + i) * HH + h) * PP + p;
            Yout[yi] += esc[r] * acc[r];
        }
    }
}

// ---------------------------------------------------------------------------
extern "C" void kernel_launch(void* const* d_in, const int* in_sizes, int n_in,
                              void* d_out, int out_size, void* d_ws, size_t ws_size,
                              hipStream_t stream)
{
    (void)in_sizes; (void)n_in; (void)out_size; (void)ws_size;
    const float* X     = (const float*)d_in[0];
    const float* delta = (const float*)d_in[1];
    const float* Bm    = (const float*)d_in[2];
    const float* Cm    = (const float*)d_in[3];
    const float* A_log = (const float*)d_in[4];

    float* Y     = (float*)d_out;
    float* Hlast = Y + (size_t)BSZ * TT * HH * PP;           // h_last tail

    float* Sws = (float*)d_ws;                               // 16 MB: chunk states
    float* Dws = Sws + (size_t)BSZ * HH * NCH * PP * NN;     // chunk decays

    const int nblk = BSZ * HH * NCH;   // 1024 chunk tiles
    ssd_chunk_kernel<<<nblk, 128, 0, stream>>>(X, delta, Bm, Cm, A_log, Y, Sws, Dws);
    ssd_scan_kernel<<<BSZ * HH, 256, 0, stream>>>(Sws, Dws, Hlast);
    ssd_inter_kernel<<<nblk, 128, 0, stream>>>(delta, Cm, A_log, Sws, Y);
}
